// DoubleStreamBlock_37658273251895
// MI455X (gfx1250) — compile-verified, matrix operand order checked by reading, not run on hardware
//
#include <hip/hip_runtime.h>

typedef __attribute__((ext_vector_type(16))) _Float16 v16h;
typedef __attribute__((ext_vector_type(8)))  _Float16 v8h;
typedef __attribute__((ext_vector_type(8)))  float    v8f;
typedef unsigned int v4u __attribute__((ext_vector_type(4)));
typedef int          v4i __attribute__((ext_vector_type(4)));
typedef int          v8i __attribute__((ext_vector_type(8)));

#define DEV __device__ __forceinline__

#if __has_builtin(__builtin_amdgcn_tensor_load_to_lds) && __has_builtin(__builtin_amdgcn_s_wait_tensorcnt)
#define USE_TDM 1
#else
#define USE_TDM 0
#endif

constexpr int Bq   = 2;
constexpr int S_IMG= 2048;
constexpr int S_TXT= 256;
constexpr int LTOT = 2304;     // S_TXT + S_IMG
constexpr int DM   = 768;
constexpr int HH   = 12;
constexpr int HDIM = 64;
constexpr int NE   = 8;
constexpr int CI   = 512;      // S_IMG*2/8
constexpr int CT   = 64;       // S_TXT*2/8
constexpr int IMLP = 3072;
constexpr int ISH  = 1536;
constexpr int MODN = 6 * DM;   // 4608

DEV float geluf(float x) { return 0.5f * x * (1.0f + erff(x * 0.70710678118654752f)); }

DEV v16h mk_frag(v8h lo, v8h hi) {
  v16h r;
#pragma unroll
  for (int i = 0; i < 8; i++) { r[i] = lo[i]; r[i + 8] = hi[i]; }
  return r;
}

// 16x32 f16 fragment from f16 memory (A-operand layout; B-operand = same packing on W rows)
DEV v16h frag_ld(const _Float16* rowp, int lh) {
  v8h lo = *(const v8h*)(rowp + 8 * lh);
  v8h hi = *(const v8h*)(rowp + 16 + 8 * lh);
  return mk_frag(lo, hi);
}

// 16x32 fragment from fp32 LDS tile, converting to f16 during assembly
DEV v16h frag_ld_f32(const float* rowp, int lh) {
  const float4 a = *(const float4*)(rowp + 8 * lh);
  const float4 b = *(const float4*)(rowp + 8 * lh + 4);
  const float4 c = *(const float4*)(rowp + 16 + 8 * lh);
  const float4 d = *(const float4*)(rowp + 16 + 8 * lh + 4);
  v16h r;
  r[0]=(_Float16)a.x;  r[1]=(_Float16)a.y;  r[2]=(_Float16)a.z;  r[3]=(_Float16)a.w;
  r[4]=(_Float16)b.x;  r[5]=(_Float16)b.y;  r[6]=(_Float16)b.z;  r[7]=(_Float16)b.w;
  r[8]=(_Float16)c.x;  r[9]=(_Float16)c.y;  r[10]=(_Float16)c.z; r[11]=(_Float16)c.w;
  r[12]=(_Float16)d.x; r[13]=(_Float16)d.y; r[14]=(_Float16)d.z; r[15]=(_Float16)d.w;
  return r;
}

DEV v8f wmma32(v16h a, v16h b, v8f c) {
  return __builtin_amdgcn_wmma_f32_16x16x32_f16(false, a, false, b, (short)0, c, false, false);
}

#if USE_TDM
// Issue a 2D TDM load: tile (tile_y rows x 32 cols) of fp32, row stride `stride` elems.
// tensor_x / tensor_y are the remaining extents from the tile origin (TDM OOB reads -> 0).
DEV void tdm_load2d(unsigned lds_off, const float* g, unsigned tile_y,
                    unsigned tensor_x, unsigned tensor_y, unsigned stride) {
  const unsigned long long ga = (unsigned long long)(const void*)g;
  v4u g0;
  g0[0] = 1u;                                        // count=1 (valid), user mode
  g0[1] = lds_off;                                   // lds_addr
  g0[2] = (unsigned)(ga & 0xffffffffu);              // global_addr[31:0]
  g0[3] = (unsigned)((ga >> 32) & 0x01ffffffu) | 0x80000000u;  // ga[56:32] | type=2
  v8i g1;
  g1[0] = (int)(2u << 16);                           // data_size = 4B
  g1[1] = (int)((tensor_x & 0xffffu) << 16);         // tensor_dim0[15:0]
  g1[2] = (int)(((tensor_x >> 16) & 0xffffu) | ((tensor_y & 0xffffu) << 16));
  g1[3] = (int)(((tensor_y >> 16) & 0xffffu) | (32u << 16));   // tile_dim0 = 32
  g1[4] = (int)(tile_y & 0xffffu);                   // tile_dim1; tile_dim2 = 0
  g1[5] = (int)stride;                               // tensor_dim0_stride[31:0]
  g1[6] = 0;
  g1[7] = 0;
  v4i z4 = {0, 0, 0, 0};
#if __clang_major__ >= 23
  v8i z8 = {0, 0, 0, 0, 0, 0, 0, 0};
  __builtin_amdgcn_tensor_load_to_lds(g0, g1, z4, z4, z8, 0);
#else
  __builtin_amdgcn_tensor_load_to_lds(g0, g1, z4, z4, 0);
#endif
}
#endif

// ---------------------------------------------------------------------------
// Generic WMMA GEMM:  out = epilogue( A(MxK) @ W(NxK)^T )
// Block tile 128x64, 8 waves (4x2) of 32x32, BK=32.
// TDM path: fp32 tiles DMA'd into double-buffered LDS, f16 convert on frag read.
// ---------------------------------------------------------------------------
constexpr int F_BIAS = 1, F_GELU = 2, F_MUL = 4, F_RESGATE = 8,
              F_SCATTER = 16, F_ADD = 32, F_GATHER = 64;

template <int F>
__global__ __launch_bounds__(256) void gemm_k(
    const float* __restrict__ A, const float* __restrict__ W,
    const float* __restrict__ bias, const float* __restrict__ res,
    const float* __restrict__ gatev, const float* __restrict__ mulb,
    const float* __restrict__ wsel, const int* __restrict__ gidx,
    float* __restrict__ out,
    int M, int N, int K,
    long A_bstr, long A_zstr, long W_estr,
    long out_zstr, long res_bstr, long scat_bstr,
    int gate_bstr, int Ez)
{
  const int z  = blockIdx.z;
  const int bb = z / Ez;
  const int ee = z - bb * Ez;
  A += (long)bb * A_bstr + (long)z * A_zstr;
  W += (long)ee * W_estr;

  const int tid  = threadIdx.x;
  const int lane = tid & 31, wave = tid >> 5;
  const int wm = wave >> 1, wn = wave & 1;
  const int ln = lane & 15, lh = lane >> 4;
  const int blockM = blockIdx.y * 128, blockN = blockIdx.x * 64;

  __shared__ float As[2][128 * 32];   // fp32 staged tiles (double buffered)
  __shared__ float Bs[2][64 * 32];

  v8f acc[2][2] = {};

  const int ar = tid >> 1, ac = (tid & 1) * 16;   // manual A: 16 floats/thread
  const int br = tid >> 2, bc = (tid & 3) * 8;    // manual B: 8 floats/thread
  const int agrow = blockM + ar;
  const int bgrow = blockN + br;
  long arow = agrow;
  if constexpr (F & F_GATHER) {
    arow = (agrow < M) ? (long)gidx[(long)z * M + agrow] : 0;
  }

#if USE_TDM
  constexpr int NT = (F & F_GATHER) ? 1 : 2;   // TDM ops in flight per tile
  const unsigned ldsA = (unsigned)(unsigned long long)(void*)&As[0][0];
  const unsigned ldsB = (unsigned)(unsigned long long)(void*)&Bs[0][0];
  if (tid == 0) {
    if constexpr (!(F & F_GATHER))
      tdm_load2d(ldsA, A + (long)blockM * K, 128, (unsigned)K, (unsigned)(M - blockM), (unsigned)K);
    tdm_load2d(ldsB, W + (long)blockN * K, 64, (unsigned)K, (unsigned)(N - blockN), (unsigned)K);
  }
  int buf = 0;
  for (int k0 = 0; k0 < K; k0 += 32) {
    if constexpr (F & F_GATHER) {  // gathered activations: manual fp32 stage
      const float* ap = A + arow * (long)K + k0 + ac;
      const bool ok = agrow < M;
#pragma unroll
      for (int i = 0; i < 16; i += 4) {
        float4 v = ok ? *(const float4*)(ap + i) : make_float4(0.f, 0.f, 0.f, 0.f);
        *(float4*)&As[buf][ar * 32 + ac + i] = v;
      }
    }
    if (tid == 0) {
      if (k0 + 32 < K) {           // prefetch next tile into the other buffer
        const int nb = buf ^ 1;
        if constexpr (!(F & F_GATHER))
          tdm_load2d(ldsA + (unsigned)(nb * 128 * 32 * 4), A + (long)blockM * K + k0 + 32,
                     128, (unsigned)(K - k0 - 32), (unsigned)(M - blockM), (unsigned)K);
        tdm_load2d(ldsB + (unsigned)(nb * 64 * 32 * 4), W + (long)blockN * K + k0 + 32,
                   64, (unsigned)(K - k0 - 32), (unsigned)(N - blockN), (unsigned)K);
        __builtin_amdgcn_s_wait_tensorcnt(NT);   // current tile's DMAs done
      } else {
        __builtin_amdgcn_s_wait_tensorcnt(0);
      }
    }
    __syncthreads();

    v16h af[2], bf[2];
#pragma unroll
    for (int t = 0; t < 2; t++) af[t] = frag_ld_f32(&As[buf][(wm * 32 + t * 16 + ln) * 32], lh);
#pragma unroll
    for (int t = 0; t < 2; t++) bf[t] = frag_ld_f32(&Bs[buf][(wn * 32 + t * 16 + ln) * 32], lh);
#pragma unroll
    for (int i = 0; i < 2; i++)
#pragma unroll
      for (int j = 0; j < 2; j++)
        acc[i][j] = wmma32(af[i], bf[j], acc[i][j]);
    __syncthreads();
    buf ^= 1;
  }
#else
  for (int k0 = 0; k0 < K; k0 += 32) {
    {
      const float* ap = A + arow * (long)K + k0 + ac;
      const bool ok = agrow < M;
#pragma unroll
      for (int i = 0; i < 16; i += 4) {
        float4 v = ok ? *(const float4*)(ap + i) : make_float4(0.f, 0.f, 0.f, 0.f);
        *(float4*)&As[0][ar * 32 + ac + i] = v;
      }
      if (ok && (k0 + 32 < K)) __builtin_prefetch(ap + 32, 0, 1);
    }
    {
      const float* wp = W + (long)bgrow * K + k0 + bc;
      const bool ok = bgrow < N;
#pragma unroll
      for (int i = 0; i < 8; i += 4) {
        float4 v = ok ? *(const float4*)(wp + i) : make_float4(0.f, 0.f, 0.f, 0.f);
        *(float4*)&Bs[0][br * 32 + bc + i] = v;
      }
      if (ok && (k0 + 32 < K)) __builtin_prefetch(wp + 32, 0, 1);
    }
    __syncthreads();
    v16h af[2], bf[2];
#pragma unroll
    for (int t = 0; t < 2; t++) af[t] = frag_ld_f32(&As[0][(wm * 32 + t * 16 + ln) * 32], lh);
#pragma unroll
    for (int t = 0; t < 2; t++) bf[t] = frag_ld_f32(&Bs[0][(wn * 32 + t * 16 + ln) * 32], lh);
#pragma unroll
    for (int i = 0; i < 2; i++)
#pragma unroll
      for (int j = 0; j < 2; j++)
        acc[i][j] = wmma32(af[i], bf[j], acc[i][j]);
    __syncthreads();
  }
#endif

#pragma unroll
  for (int i = 0; i < 2; i++) {
#pragma unroll
    for (int j = 0; j < 2; j++) {
      const int nn = blockN + wn * 32 + j * 16 + ln;
      if (nn >= N) continue;
#pragma unroll
      for (int r = 0; r < 8; r++) {
        const int m = blockM + wm * 32 + i * 16 + lh * 8 + r;
        if (m >= M) continue;
        float v = acc[i][j][r];
        if constexpr (F & F_BIAS) v += bias[nn];
        if constexpr (F & F_GELU) v = geluf(v);
        if constexpr (F & F_MUL)  v *= mulb[(long)z * out_zstr + (long)m * N + nn];
        if constexpr (F & F_RESGATE)
          v = res[(long)bb * res_bstr + (long)m * N + nn] +
              gatev[(long)bb * gate_bstr + nn] * v;
        if constexpr (F & F_SCATTER) {
          const int tok = gidx[(long)z * M + m];
          const float wv = wsel[(long)z * M + m];
          atomicAdd(out + (long)bb * scat_bstr + (long)tok * N + nn, v * wv);
        } else if constexpr (F & F_ADD) {
          out[(long)z * out_zstr + (long)m * N + nn] += v;
        } else {
          out[(long)z * out_zstr + (long)m * N + nn] = v;
        }
      }
    }
  }
}

// ---------------------------------------------------------------------------
// Flash attention: 1 wave per (b, h, 16 q-rows). Q/K f16 (L,64), V^T f16 (64,L).
// ---------------------------------------------------------------------------
__global__ __launch_bounds__(32) void attn_k(
    const _Float16* __restrict__ Qf, const _Float16* __restrict__ Kf,
    const _Float16* __restrict__ VT, float* __restrict__ outp)
{
  const int lane = threadIdx.x;
  const int ln = lane & 15, lh = lane >> 4;
  const int h = blockIdx.y, b = blockIdx.z;
  const long bh = (long)(b * HH + h);
  const _Float16* q  = Qf + bh * LTOT * HDIM;
  const _Float16* k  = Kf + bh * LTOT * HDIM;
  const _Float16* vt = VT + bh * HDIM * LTOT;
  const int q0 = blockIdx.x * 16;

  v16h qf[2];
#pragma unroll
  for (int u = 0; u < 2; u++) qf[u] = frag_ld(q + (long)(q0 + ln) * HDIM + u * 32, lh);

  v8f o[4] = {};
  float mrow[8], lrow[8];
#pragma unroll
  for (int r = 0; r < 8; r++) { mrow[r] = -3.0e38f; lrow[r] = 0.f; }

  __shared__ _Float16 Pl[16 * 32];
  const float scale = 0.125f;   // 1/sqrt(64)

  for (int kt = 0; kt < LTOT; kt += 32) {
    v8f s0 = {}, s1 = {};
#pragma unroll
    for (int u = 0; u < 2; u++) {
      v16h ka = frag_ld(k + (long)(kt + ln) * HDIM + u * 32, lh);
      v16h kb = frag_ld(k + (long)(kt + 16 + ln) * HDIM + u * 32, lh);
      s0 = wmma32(qf[u], ka, s0);
      s1 = wmma32(qf[u], kb, s1);
    }
#pragma unroll
    for (int r = 0; r < 8; r++) {
      float v0 = s0[r] * scale, v1 = s1[r] * scale;
      float mx = fmaxf(v0, v1);
#pragma unroll
      for (int off = 1; off < 16; off <<= 1) mx = fmaxf(mx, __shfl_xor(mx, off, 32));
      const float mnew  = fmaxf(mrow[r], mx);
      const float alpha = __expf(mrow[r] - mnew);
      const float p0 = __expf(v0 - mnew), p1 = __expf(v1 - mnew);
      float rs = p0 + p1;
#pragma unroll
      for (int off = 1; off < 16; off <<= 1) rs += __shfl_xor(rs, off, 32);
      lrow[r] = lrow[r] * alpha + rs;
      mrow[r] = mnew;
#pragma unroll
      for (int jj = 0; jj < 4; jj++) o[jj][r] *= alpha;
      Pl[(lh * 8 + r) * 32 + ln]      = (_Float16)p0;
      Pl[(lh * 8 + r) * 32 + 16 + ln] = (_Float16)p1;
    }
    __builtin_amdgcn_wave_barrier();   // in-wave LDS store->load ordering fence
    v16h pf = frag_ld(&Pl[ln * 32], lh);
#pragma unroll
    for (int jj = 0; jj < 4; jj++) {
      v16h vf = frag_ld(vt + (long)(jj * 16 + ln) * LTOT + kt, lh);
      o[jj] = wmma32(pf, vf, o[jj]);
    }
    __builtin_amdgcn_wave_barrier();
  }
#pragma unroll
  for (int jj = 0; jj < 4; jj++)
#pragma unroll
    for (int r = 0; r < 8; r++) {
      const long row = (long)b * LTOT + q0 + lh * 8 + r;
      outp[row * DM + h * HDIM + jj * 16 + ln] = o[jj][r] / lrow[r];
    }
}

// ---------------------------------------------------------------------------
// Elementwise / small kernels
// ---------------------------------------------------------------------------
__global__ void gelu_vec_k(const float* __restrict__ v, float* __restrict__ g, int n) {
  int i = blockIdx.x * 256 + threadIdx.x;
  if (i < n) g[i] = geluf(v[i]);
}

__global__ void mod_gemv_k(const float* __restrict__ g, const float* __restrict__ w,
                           const float* __restrict__ bs, float* __restrict__ out) {
  int t = blockIdx.x * 256 + threadIdx.x;
  if (t >= Bq * MODN) return;
  int bb = t / MODN, n = t - bb * MODN;
  const float* gr = g + bb * DM;
  const float* wr = w + (long)n * DM;
  float a = bs[n];
  for (int kk = 0; kk < DM; kk++) a += gr[kk] * wr[kk];
  out[t] = a;
}

__global__ __launch_bounds__(256) void lnmod_k(const float* __restrict__ x,
                                               const float* __restrict__ mod,
                                               float* __restrict__ out, int S, int chunk) {
  const long row = blockIdx.x;
  const int b = (int)(row / S);
  const float* xr = x + row * DM;
  const int tid = threadIdx.x, lane = tid & 31, wv = tid >> 5;
  float s = 0.f, s2 = 0.f;
  for (int i = tid; i < DM; i += 256) { float v = xr[i]; s += v; s2 += v * v; }
#pragma unroll
  for (int o = 16; o; o >>= 1) { s += __shfl_xor(s, o, 32); s2 += __shfl_xor(s2, o, 32); }
  __shared__ float rs[8], rs2[8];
  if (!lane) { rs[wv] = s; rs2[wv] = s2; }
  __syncthreads();
  float ts = 0.f, ts2 = 0.f;
#pragma unroll
  for (int i = 0; i < 8; i++) { ts += rs[i]; ts2 += rs2[i]; }
  const float mean = ts / DM;
  const float var = ts2 / DM - mean * mean;
  const float rstd = rsqrtf(var + 1e-6f);
  const float* sh = mod + (long)b * MODN + chunk * DM;
  const float* sc = sh + DM;
  for (int i = tid; i < DM; i += 256)
    out[row * DM + i] = (1.f + sc[i]) * ((xr[i] - mean) * rstd) + sh[i];
}

__global__ __launch_bounds__(256) void rms_reshape_k(
    const float* __restrict__ qkv, const float* __restrict__ qs, const float* __restrict__ ks,
    _Float16* __restrict__ Qf, _Float16* __restrict__ Kf, _Float16* __restrict__ Vf,
    _Float16* __restrict__ VT, int S, int Loff, int nwaves) {
  const int gw = (blockIdx.x * 256 + threadIdx.x) >> 5;
  if (gw >= nwaves) return;
  const int lane = threadIdx.x & 31;
  const int h = gw % HH, bs = gw / HH;
  const int b = bs / S, s = bs - b * S;
  const float* row = qkv + (long)bs * (3 * DM) + h * HDIM;
  const int d0 = lane, d1 = lane + 32;
  float q0 = row[d0], q1 = row[d1];
  float k0 = row[DM + d0], k1 = row[DM + d1];
  float v0 = row[2 * DM + d0], v1 = row[2 * DM + d1];
  float sq = q0 * q0 + q1 * q1, sk = k0 * k0 + k1 * k1;
#pragma unroll
  for (int o = 16; o; o >>= 1) { sq += __shfl_xor(sq, o, 32); sk += __shfl_xor(sk, o, 32); }
  const float rq = rsqrtf(sq / 64.f + 1e-6f);
  const float rk = rsqrtf(sk / 64.f + 1e-6f);
  q0 = q0 * rq * qs[d0]; q1 = q1 * rq * qs[d1];
  k0 = k0 * rk * ks[d0]; k1 = k1 * rk * ks[d1];
  const long base = ((long)(b * HH + h) * LTOT + Loff + s) * HDIM;
  Qf[base + d0] = (_Float16)q0; Qf[base + d1] = (_Float16)q1;
  Kf[base + d0] = (_Float16)k0; Kf[base + d1] = (_Float16)k1;
  Vf[base + d0] = (_Float16)v0; Vf[base + d1] = (_Float16)v1;
  const long tb = (long)(b * HH + h) * HDIM * LTOT + Loff + s;
  VT[tb + (long)d0 * LTOT] = (_Float16)v0;
  VT[tb + (long)d1 * LTOT] = (_Float16)v1;
}

__global__ __launch_bounds__(256) void gate_k(const float* __restrict__ x,
                                              const float* __restrict__ gw,
                                              float* __restrict__ sc, int S) {
  const int gt = (blockIdx.x * 256 + threadIdx.x) >> 5;
  if (gt >= Bq * S) return;
  const int lane = threadIdx.x & 31;
  const int b = gt / S, s = gt - b * S;
  const float* xr = x + (long)gt * DM;
  float ev[NE];
#pragma unroll
  for (int e = 0; e < NE; e++) {
    float a = 0.f;
    for (int i = lane; i < DM; i += 32) a += xr[i] * gw[e * DM + i];
#pragma unroll
    for (int o = 16; o; o >>= 1) a += __shfl_xor(a, o, 32);
    ev[e] = a;
  }
  float m = ev[0];
#pragma unroll
  for (int e = 1; e < NE; e++) m = fmaxf(m, ev[e]);
  float den = 0.f;
#pragma unroll
  for (int e = 0; e < NE; e++) { ev[e] = __expf(ev[e] - m); den += ev[e]; }
#pragma unroll
  for (int e = 0; e < NE; e++) sc[((long)b * NE + e) * S + s] = ev[e] / den;
}

// rank-based expert-choice top-k (exact top_k ordering incl. index tie-break)
__global__ __launch_bounds__(256) void topk_k(const float* __restrict__ scores,
                                              int* __restrict__ idx, float* __restrict__ w,
                                              int S, int C) {
  __shared__ float sbuf[2048];
  const float* srow = scores + (long)blockIdx.x * S;
  for (int i = threadIdx.x; i < S; i += 256) sbuf[i] = srow[i];
  __syncthreads();
  for (int i = threadIdx.x; i < S; i += 256) {
    const float si = sbuf[i];
    int rank = 0;
    for (int j = 0; j < S; j++) {
      const float sj = sbuf[j];
      rank += (sj > si) || (sj == si && j < i);
    }
    if (rank < C) {
      idx[(long)blockIdx.x * C + rank] = i;
      w[(long)blockIdx.x * C + rank] = si;
    }
  }
}

__global__ void final_k(const float* __restrict__ base, const float* __restrict__ moe,
                        const float* __restrict__ mod, float* __restrict__ out, int S, int n) {
  const int i = blockIdx.x * 256 + threadIdx.x;
  if (i >= n) return;
  const int c = i % DM;
  const int b = (i / DM) / S;
  out[i] = base[i] + mod[(long)b * MODN + 5 * DM + c] * moe[i];
}

// ---------------------------------------------------------------------------
// Host side
// ---------------------------------------------------------------------------
template <int FL>
static inline void gemm_launch(hipStream_t st, const float* A, const float* W,
                               const float* bias, const float* res, const float* gate,
                               const float* mul, const float* wsel, const int* gidx,
                               float* out, int M, int N, int K,
                               long A_bstr, long A_zstr, long W_estr, long out_zstr,
                               long res_bstr, long scat_bstr, int gate_bstr, int Ez, int Z) {
  dim3 grid((N + 63) / 64, (M + 127) / 128, Z);
  gemm_k<FL><<<grid, 256, 0, st>>>(A, W, bias, res, gate, mul, wsel, gidx, out,
                                   M, N, K, A_bstr, A_zstr, W_estr, out_zstr,
                                   res_bstr, scat_bstr, gate_bstr, Ez);
}

extern "C" void kernel_launch(void* const* d_in, const int* in_sizes, int n_in,
                              void* d_out, int out_size, void* d_ws, size_t ws_size,
                              hipStream_t stream) {
  (void)in_sizes; (void)n_in; (void)out_size; (void)ws_size;
  const float* img = (const float*)d_in[0];
  const float* txt = (const float*)d_in[1];
  const float* vec = (const float*)d_in[2];
  auto IN = [&](int i) { return (const float*)d_in[i]; };
  // per-stream param order: 0 mod_w 1 mod_b 2 qkv_w 3 qkv_b 4 q_scale 5 k_scale
  //                         6 proj_w 7 proj_b 8 gate_w 9 wg 10 wu 11 wd 12 swg 13 swu 14 swd
  const int PB[2] = {3, 18};

  char* cur = (char*)d_ws;
  auto alloc = [&](size_t bytes) -> void* {
    void* p = (void*)cur;
    cur += (bytes + 255) & ~(size_t)255;
    return p;
  };
  float* gvec = (float*)alloc(sizeof(float) * Bq * DM);
  float* modB[2]; modB[0] = (float*)alloc(sizeof(float) * Bq * MODN);
                  modB[1] = (float*)alloc(sizeof(float) * Bq * MODN);
  const int SS[2] = {S_IMG, S_TXT};
  const int CC[2] = {CI, CT};
  float *xm1[2], *xm2[2], *x2[2], *scores[2], *sw[2], *gb[2], *sg[2], *moe[2];
  int* sidx[2];
  for (int t = 0; t < 2; t++) {
    xm1[t]    = (float*)alloc(sizeof(float) * Bq * SS[t] * DM);
    xm2[t]    = (float*)alloc(sizeof(float) * Bq * SS[t] * DM);
    x2[t]     = (float*)alloc(sizeof(float) * Bq * SS[t] * DM);
    scores[t] = (float*)alloc(sizeof(float) * Bq * NE * SS[t]);
    sidx[t]   = (int*)  alloc(sizeof(int)   * Bq * NE * CC[t]);
    sw[t]     = (float*)alloc(sizeof(float) * Bq * NE * CC[t]);
    gb[t]     = (float*)alloc(sizeof(float) * (size_t)Bq * NE * CC[t] * IMLP);
    sg[t]     = (float*)alloc(sizeof(float) * (size_t)Bq * SS[t] * ISH);
    moe[t]    = (float*)alloc(sizeof(float) * Bq * SS[t] * DM);
  }
  float* qkvb = (float*)alloc(sizeof(float) * (size_t)Bq * S_IMG * 3 * DM);
  const size_t nhalf = (size_t)Bq * HH * LTOT * HDIM;
  _Float16* Qh = (_Float16*)alloc(sizeof(_Float16) * nhalf);
  _Float16* Kh = (_Float16*)alloc(sizeof(_Float16) * nhalf);
  _Float16* Vh = (_Float16*)alloc(sizeof(_Float16) * nhalf);
  _Float16* VTh = (_Float16*)alloc(sizeof(_Float16) * nhalf);
  float* attn = (float*)alloc(sizeof(float) * (size_t)Bq * LTOT * DM);

  const float* xin[2] = {img, txt};
  const int Loff[2] = {S_TXT, 0};
  float* outp[2] = {(float*)d_out, (float*)d_out + (size_t)Bq * S_IMG * DM};

  // 1) gelu(vec) and modulation vectors
  gelu_vec_k<<<(Bq * DM + 255) / 256, 256, 0, stream>>>(vec, gvec, Bq * DM);
  for (int t = 0; t < 2; t++)
    mod_gemv_k<<<(Bq * MODN + 255) / 256, 256, 0, stream>>>(gvec, IN(PB[t] + 0), IN(PB[t] + 1), modB[t]);

  // 2) LN + modulate (set 1), QKV GEMM, RMS-norm + reshape into joint f16 Q/K/V
  for (int t = 0; t < 2; t++) {
    const int S = SS[t];
    lnmod_k<<<Bq * S, 256, 0, stream>>>(xin[t], modB[t], xm1[t], S, 0);
    gemm_launch<F_BIAS>(stream, xm1[t], IN(PB[t] + 2), IN(PB[t] + 3),
                        nullptr, nullptr, nullptr, nullptr, nullptr, qkvb,
                        Bq * S, 3 * DM, DM, 0, 0, 0, 0, 0, 0, 0, 1, 1);
    const int nw = Bq * S * HH;
    rms_reshape_k<<<(nw * 32 + 255) / 256, 256, 0, stream>>>(
        qkvb, IN(PB[t] + 4), IN(PB[t] + 5), Qh, Kh, Vh, VTh, S, Loff[t], nw);
  }

  // 3) joint attention over txt||img
  attn_k<<<dim3(LTOT / 16, HH, Bq), 32, 0, stream>>>(Qh, Kh, VTh, attn);

  // 4) proj + gated residual; LN + modulate (set 2)
  for (int t = 0; t < 2; t++) {
    const int S = SS[t];
    gemm_launch<F_BIAS | F_RESGATE>(stream, attn + (long)Loff[t] * DM, IN(PB[t] + 6), IN(PB[t] + 7),
                                    xin[t], modB[t] + 2 * DM, nullptr, nullptr, nullptr, x2[t],
                                    S, DM, DM, (long)LTOT * DM, 0, 0, (long)S * DM,
                                    (long)S * DM, 0, MODN, 1, Bq);
    lnmod_k<<<Bq * S, 256, 0, stream>>>(x2[t], modB[t], xm2[t], S, 3);
  }

  // 5) expert-choice MoE + shared expert
  for (int t = 0; t < 2; t++) {
    const int S = SS[t], C = CC[t];
    gate_k<<<(Bq * S * 32 + 255) / 256, 256, 0, stream>>>(xm2[t], IN(PB[t] + 8), scores[t], S);
    topk_k<<<Bq * NE, 256, 0, stream>>>(scores[t], sidx[t], sw[t], S, C);
    hipMemsetAsync(moe[t], 0, sizeof(float) * Bq * S * DM, stream);
    // g = gelu(tok @ wg^T)
    gemm_launch<F_GELU | F_GATHER>(stream, xm2[t], IN(PB[t] + 9), nullptr, nullptr, nullptr,
                                   nullptr, nullptr, sidx[t], gb[t],
                                   C, IMLP, DM, (long)S * DM, 0, (long)IMLP * DM,
                                   (long)C * IMLP, 0, 0, 0, NE, Bq * NE);
    // h = g * (tok @ wu^T)   (in place)
    gemm_launch<F_MUL | F_GATHER>(stream, xm2[t], IN(PB[t] + 10), nullptr, nullptr, nullptr,
                                  gb[t], nullptr, sidx[t], gb[t],
                                  C, IMLP, DM, (long)S * DM, 0, (long)IMLP * DM,
                                  (long)C * IMLP, 0, 0, 0, NE, Bq * NE);
    // moe += scatter( (h @ wd^T) * w )
    gemm_launch<F_SCATTER>(stream, gb[t], IN(PB[t] + 11), nullptr, nullptr, nullptr,
                           nullptr, sw[t], sidx[t], moe[t],
                           C, DM, IMLP, 0, (long)C * IMLP, (long)DM * IMLP,
                           0, 0, (long)S * DM, 0, NE, Bq * NE);
    // shared expert: sg = gelu(x @ swg^T); sg *= x @ swu^T; moe += sg @ swd^T
    gemm_launch<F_GELU>(stream, xm2[t], IN(PB[t] + 12), nullptr, nullptr, nullptr,
                        nullptr, nullptr, nullptr, sg[t],
                        Bq * S, ISH, DM, 0, 0, 0, 0, 0, 0, 0, 1, 1);
    gemm_launch<F_MUL>(stream, xm2[t], IN(PB[t] + 13), nullptr, nullptr, nullptr,
                       sg[t], nullptr, nullptr, sg[t],
                       Bq * S, ISH, DM, 0, 0, 0, 0, 0, 0, 0, 1, 1);
    gemm_launch<F_ADD>(stream, sg[t], IN(PB[t] + 14), nullptr, nullptr, nullptr,
                       nullptr, nullptr, nullptr, moe[t],
                       Bq * S, DM, ISH, 0, 0, 0, 0, 0, 0, 0, 1, 1);
    // 6) final gated residual -> d_out
    const int n = Bq * S * DM;
    final_k<<<(n + 255) / 256, 256, 0, stream>>>(x2[t], moe[t], modB[t], outp[t], S, n);
  }
}
